// BondBackbone_3332894622337
// MI455X (gfx1250) — matrix operand order repaired and checked by reading, not compiled, not run on hardware
//
#include <hip/hip_runtime.h>
#include <hip/hip_bf16.h>
#include <stdint.h>

typedef __attribute__((ext_vector_type(16))) _Float16 v16h;
typedef __attribute__((ext_vector_type(8)))  _Float16 v8h;
typedef __attribute__((ext_vector_type(8)))  float    v8f;
typedef int v4i __attribute__((vector_size(16)));

#define IN_DIM   140
#define K_PAD    448      // 3*IN_DIM=420 padded to multiple of 32
#define OUT_DIM  128
#define N_ISS    5000
#define N_SEC    25

// CDNA5 async global->LDS copies (ASYNCcnt-tracked), with safe fallback.
#if defined(__AMDGCN__) && __has_builtin(__builtin_amdgcn_global_load_async_to_lds_b128) && __has_builtin(__builtin_amdgcn_s_wait_asynccnt)
#define USE_ASYNC 1
#else
#define USE_ASYNC 0
#endif

__device__ __forceinline__ void cp16(_Float16* dst_lds, const _Float16* src_glb) {
#if USE_ASYNC
    __builtin_amdgcn_global_load_async_to_lds_b128(
        (__attribute__((address_space(1))) v4i*)(uintptr_t)src_glb,
        (__attribute__((address_space(3))) v4i*)(uint32_t)(uintptr_t)dst_lds,
        0, 0);
#else
    *(uint4*)dst_lds = *(const uint4*)src_glb;
#endif
}

__device__ __forceinline__ void async_wait() {
#if USE_ASYNC
    __builtin_amdgcn_s_wait_asynccnt(0);   // s_wait_asynccnt 0
#endif
}

// ---------------------------------------------------------------- zero
__global__ void k_zero(float* __restrict__ p, int n) {
    int i = blockIdx.x * blockDim.x + threadIdx.x;
    if (i < n) p[i] = 0.f;
}

// ---------------------------------------------------------------- h_self build (1 row / 64-thread block)
__global__ __launch_bounds__(64) void k_build_hself(
    const int* __restrict__ node_ids, const int* __restrict__ cat_rating,
    const int* __restrict__ cat_currency, const float* __restrict__ nums,
    const int* __restrict__ n2i, const int* __restrict__ n2s,
    const float* __restrict__ id_emb, const float* __restrict__ rating_emb,
    const float* __restrict__ currency_emb,
    const float* __restrict__ W1, const float* __restrict__ b1,
    const float* __restrict__ W2, const float* __restrict__ b2,
    float* __restrict__ h_self, int* __restrict__ issuers, int* __restrict__ sectors,
    float* __restrict__ iss_sums, float* __restrict__ iss_cnt)
{
    __shared__ float sn[16];
    __shared__ float sh1[64];
    __shared__ float srow[IN_DIM];
    const int row = blockIdx.x;
    const int j   = threadIdx.x;
    const int nid = node_ids[row];

    if (j < 16) sn[j] = nums[row * 16 + j];
    __syncthreads();

    float a1 = b1[j];
    #pragma unroll
    for (int k = 0; k < 16; ++k) a1 = fmaf(sn[k], W1[k * 64 + j], a1);
    sh1[j] = a1 > 0.f ? a1 : 0.f;
    __syncthreads();

    float a2 = b2[j];
    #pragma unroll 8
    for (int k = 0; k < 64; ++k) a2 = fmaf(sh1[k], W2[k * 64 + j], a2);
    a2 = a2 > 0.f ? a2 : 0.f;

    srow[j] = id_emb[(size_t)nid * 64 + j];                            // [0,64)
    if (j < 8) srow[64 + j] = rating_emb[cat_rating[row] * 8 + j];     // [64,72)
    if (j < 4) srow[72 + j] = currency_emb[cat_currency[row] * 4 + j]; // [72,76)
    srow[76 + j] = a2;                                                 // [76,140)
    __syncthreads();

    const int iss = n2i[nid];
    const int sec = n2s[nid];
    if (j == 0) {
        issuers[row] = iss;
        sectors[row] = sec;
        atomicAdd(&iss_cnt[iss], 1.0f);
    }
    for (int c = j; c < IN_DIM; c += 64) {
        float v = srow[c];
        h_self[(size_t)row * IN_DIM + c] = v;
        atomicAdd(&iss_sums[iss * IN_DIM + c], v);  // 5000 segments: spread, global atomics OK
    }
}

// ---------------------------------------------------------------- sector sums: LDS-hierarchical (25 segments)
__global__ __launch_bounds__(256) void k_sector_sums(
    const float* __restrict__ h_self, const int* __restrict__ sectors,
    float* __restrict__ sec_sums, float* __restrict__ sec_cnt, int rows_per_block)
{
    __shared__ float ssum[N_SEC * IN_DIM];
    __shared__ float scnt[N_SEC];
    const int tid = threadIdx.x;
    for (int i = tid; i < N_SEC * IN_DIM; i += 256) ssum[i] = 0.f;
    if (tid < N_SEC) scnt[tid] = 0.f;
    __syncthreads();

    const int r0 = blockIdx.x * rows_per_block;
    for (int idx = tid; idx < rows_per_block * IN_DIM; idx += 256) {
        int r = r0 + idx / IN_DIM;
        int c = idx % IN_DIM;
        int s = sectors[r];
        atomicAdd(&ssum[s * IN_DIM + c], h_self[(size_t)r * IN_DIM + c]);  // ds_add_f32
        if (c == 0) atomicAdd(&scnt[s], 1.0f);
    }
    __syncthreads();
    for (int i = tid; i < N_SEC * IN_DIM; i += 256) atomicAdd(&sec_sums[i], ssum[i]);
    if (tid < N_SEC) atomicAdd(&sec_cnt[tid], scnt[tid]);
}

// ---------------------------------------------------------------- sums -> means (in place)
__global__ void k_finalize(float* __restrict__ iss_sums, const float* __restrict__ iss_cnt,
                           float* __restrict__ sec_sums, const float* __restrict__ sec_cnt)
{
    int i = blockIdx.x * blockDim.x + threadIdx.x;
    if (i < N_ISS * IN_DIM) {
        iss_sums[i] /= fmaxf(iss_cnt[i / IN_DIM], 1.f);
    } else {
        int k = i - N_ISS * IN_DIM;
        if (k < N_SEC * IN_DIM) sec_sums[k] /= fmaxf(sec_cnt[k / IN_DIM], 1.f);
    }
}

// ---------------------------------------------------------------- fp32 weights -> f16, TRANSPOSED [n][k], K 420->448
// W1t[n*448 + k] = agg_W1[k*128 + n] (0 for k>=420);  W2t[n*128 + k] = agg_W2[k*128 + n]
// [n][k] layout makes every lane's B fragment a contiguous 32B span -> 2x ds_load_b128.
__global__ void k_cvt_weights_t(const float* __restrict__ agg_W1, const float* __restrict__ agg_W2,
                                _Float16* __restrict__ W1t, _Float16* __restrict__ W2t)
{
    int i = blockIdx.x * blockDim.x + threadIdx.x;
    if (i < OUT_DIM * K_PAD) {
        int n = i / K_PAD, k = i - n * K_PAD;
        W1t[i] = (_Float16)(k < 420 ? agg_W1[k * OUT_DIM + n] : 0.f);
    } else {
        int j = i - OUT_DIM * K_PAD;
        if (j < OUT_DIM * OUT_DIM) {
            int n = j >> 7, k = j & 127;
            W2t[j] = (_Float16)agg_W2[k * OUT_DIM + n];
        }
    }
}

// ---------------------------------------------------------------- fused head:
//   X = concat(h_self, iss_mean, sec_mean, 0pad) built directly in LDS (saves ~29MB HBM round-trip)
//   out = relu(X[16,448] @ W1 + b1) @ W2 + b2  via v_wmma_f32_16x16x32_f16
// Block = 256 threads (8 waves); wave w owns output columns [16w,16w+16).
// Layouts used: A lane: half=lane>>4, m=lane&15, k=16*half+s, this code's order, B same (2x b128)
//                        B (from [n][k] LDS): 32B contiguous per lane (2x b128)
//                        C/D: m = vgpr + 8*half, n = lane&15
__global__ __launch_bounds__(256) void k_agg_gemm(
    const float* __restrict__ h_self, const int* __restrict__ issuers,
    const int* __restrict__ sectors, const float* __restrict__ iss_means,
    const float* __restrict__ sec_means,
    const _Float16* __restrict__ W1t,  // [128][448]
    const float*    __restrict__ b1,
    const _Float16* __restrict__ W2t,  // [128][128]
    const float*    __restrict__ b2,
    float*          __restrict__ out)  // [B,128]
{
    __shared__ _Float16 sX  [16 * K_PAD];           // 14 KB, row-major
    __shared__ _Float16 sWt [2][32 * OUT_DIM];      // 2x 8 KB ping-pong, layout [n*32 + k]
    __shared__ _Float16 sW2t[OUT_DIM * OUT_DIM];    // 32 KB, layout [n*128 + k]
    __shared__ _Float16 sY  [16 * OUT_DIM];         // 4 KB
    __shared__ int sIss[16], sSec[16];

    const int tid  = threadIdx.x;
    const int wv   = tid >> 5;
    const int lane = tid & 31;
    const int half = lane >> 4;
    const int mn   = lane & 15;
    const int col  = wv * 16 + mn;
    const int r0   = blockIdx.x * 16;

    if (tid < 16) { sIss[tid] = issuers[r0 + tid]; sSec[tid] = sectors[r0 + tid]; }

    // stage full W2t (2048 x 16B chunks, contiguous) — async if available
    for (int c = tid; c < OUT_DIM * OUT_DIM / 8; c += 256)
        cp16(&sW2t[c * 8], &W2t[c * 8]);

    // stage W1 slab 0 into ping buffer: 128 cols x 32 k = 512 x 16B chunks
    for (int c = tid; c < 512; c += 256) {
        int n = c >> 2, p = c & 3;
        cp16(&sWt[0][n * 32 + p * 8], &W1t[n * K_PAD + 0 * 32 + p * 8]);
    }

    __syncthreads();   // sIss/sSec visible for the X build

    // build X tile straight into LDS (f16), gathering the two pooled means
    for (int idx = tid; idx < 16 * K_PAD; idx += 256) {
        int r = idx / K_PAD, c = idx - r * K_PAD;
        float v = 0.f;
        if (c < IN_DIM)          v = h_self[(size_t)(r0 + r) * IN_DIM + c];
        else if (c < 2 * IN_DIM) v = iss_means[sIss[r] * IN_DIM + (c - IN_DIM)];
        else if (c < 3 * IN_DIM) v = sec_means[sSec[r] * IN_DIM + (c - 2 * IN_DIM)];
        sX[idx] = (_Float16)v;
    }

    v8f acc = {};
    for (int kt = 0; kt < K_PAD / 32; ++kt) {
        async_wait();           // my async LDS writes done
        __syncthreads();        // everyone's slab kt (and X, W2t) visible

        if (kt + 1 < K_PAD / 32) {          // overlap: stage slab kt+1 into the other buffer
            const _Float16* src = W1t + (kt + 1) * 32;
            _Float16* dst = sWt[(kt + 1) & 1];
            for (int c = tid; c < 512; c += 256) {
                int n = c >> 2, p = c & 3;
                cp16(&dst[n * 32 + p * 8], &src[n * K_PAD + p * 8]);
            }
            __builtin_prefetch(W1t + (kt + 2) * 32, 0, 0);  // warm GL2 for slab after next
        }

        // A fragment: two contiguous 16B LDS loads
        const _Float16* ax = &sX[mn * K_PAD + kt * 32 + half * 16];
        v8h a0 = *(const v8h*)ax;
        v8h a1 = *(const v8h*)(ax + 8);
        v16h a = __builtin_shufflevector(a0, a1, 0,1,2,3,4,5,6,7,8,9,10,11,12,13,14,15);
        // B fragment: one contiguous 32B LDS span
        const _Float16* bx = &sWt[kt & 1][col * 32 + half * 16];
        v8h b0 = *(const v8h*)bx;
        v8h b1v = *(const v8h*)(bx + 8);
        v16h b = __builtin_shufflevector(b0, b1v, 0,1,2,3,4,5,6,7,8,9,10,11,12,13,14,15);

        acc = __builtin_amdgcn_wmma_f32_16x16x32_f16(
            false, a, false, b, (short)0, acc, false, false);
    }

    // bias + relu -> f16 Y tile in LDS
    __syncthreads();
    {
        const float bias = b1[col];
        #pragma unroll
        for (int i = 0; i < 8; ++i) {
            int m = i + half * 8;
            float v = acc[i] + bias;
            sY[m * OUT_DIM + col] = (_Float16)(v > 0.f ? v : 0.f);
        }
    }
    __syncthreads();

    v8f acc2 = {};
    #pragma unroll
    for (int kt = 0; kt < OUT_DIM / 32; ++kt) {
        const _Float16* ay = &sY[mn * OUT_DIM + kt * 32 + half * 16];
        v8h a0 = *(const v8h*)ay;
        v8h a1 = *(const v8h*)(ay + 8);
        v16h a = __builtin_shufflevector(a0, a1, 0,1,2,3,4,5,6,7,8,9,10,11,12,13,14,15);
        const _Float16* bx = &sW2t[col * OUT_DIM + kt * 32 + half * 16];
        v8h b0 = *(const v8h*)bx;
        v8h b1v = *(const v8h*)(bx + 8);
        v16h b = __builtin_shufflevector(b0, b1v, 0,1,2,3,4,5,6,7,8,9,10,11,12,13,14,15);

        acc2 = __builtin_amdgcn_wmma_f32_16x16x32_f16(
            false, a, false, b, (short)0, acc2, false, false);
    }

    {
        const float bias = b2[col];
        #pragma unroll
        for (int i = 0; i < 8; ++i) {
            int m = i + half * 8;
            out[(size_t)(r0 + m) * OUT_DIM + col] = acc2[i] + bias;
        }
    }
}

// ---------------------------------------------------------------- host
extern "C" void kernel_launch(void* const* d_in, const int* in_sizes, int n_in,
                              void* d_out, int out_size, void* d_ws, size_t ws_size,
                              hipStream_t stream)
{
    const int*   node_ids     = (const int*)  d_in[0];
    const int*   cat_rating   = (const int*)  d_in[1];
    const int*   cat_currency = (const int*)  d_in[2];
    const float* nums         = (const float*)d_in[3];
    const int*   n2i          = (const int*)  d_in[4];
    const int*   n2s          = (const int*)  d_in[5];
    const float* id_emb       = (const float*)d_in[6];
    const float* rating_emb   = (const float*)d_in[7];
    const float* currency_emb = (const float*)d_in[8];
    const float* num_W1       = (const float*)d_in[9];
    const float* num_b1       = (const float*)d_in[10];
    const float* num_W2       = (const float*)d_in[11];
    const float* num_b2       = (const float*)d_in[12];
    const float* agg_W1       = (const float*)d_in[13];
    const float* agg_b1       = (const float*)d_in[14];
    const float* agg_W2       = (const float*)d_in[15];
    const float* agg_b2       = (const float*)d_in[16];
    float* out = (float*)d_out;

    const int B = in_sizes[0];   // 16384

    // workspace carve-up (256B-aligned)
    char* ws = (char*)d_ws;
    size_t off = 0;
    auto take = [&](size_t bytes) -> char* {
        off = (off + 255) & ~(size_t)255;
        char* p = ws + off;
        off += bytes;
        return p;
    };
    float* h_self = (float*)take((size_t)B * IN_DIM * 4);
    const int nzero = N_ISS * IN_DIM + N_SEC * IN_DIM + N_ISS + N_SEC;
    float* zero_base = (float*)take((size_t)nzero * 4);
    float* iss_sums = zero_base;
    float* sec_sums = iss_sums + N_ISS * IN_DIM;
    float* iss_cnt  = sec_sums + N_SEC * IN_DIM;
    float* sec_cnt  = iss_cnt  + N_ISS;
    int* issuers = (int*)take((size_t)B * 4);
    int* sectors = (int*)take((size_t)B * 4);
    _Float16* W1t = (_Float16*)take((size_t)OUT_DIM * K_PAD * 2 + 256); // +256: prefetch overshoot pad
    _Float16* W2t = (_Float16*)take((size_t)OUT_DIM * OUT_DIM * 2);

    k_zero<<<(nzero + 255) / 256, 256, 0, stream>>>(zero_base, nzero);

    k_build_hself<<<B, 64, 0, stream>>>(
        node_ids, cat_rating, cat_currency, nums, n2i, n2s,
        id_emb, rating_emb, currency_emb,
        num_W1, num_b1, num_W2, num_b2,
        h_self, issuers, sectors, iss_sums, iss_cnt);

    const int rows_per_block = 256;
    k_sector_sums<<<B / rows_per_block, 256, 0, stream>>>(
        h_self, sectors, sec_sums, sec_cnt, rows_per_block);

    k_finalize<<<((N_ISS + N_SEC) * IN_DIM + 255) / 256, 256, 0, stream>>>(
        iss_sums, iss_cnt, sec_sums, sec_cnt);

    k_cvt_weights_t<<<(OUT_DIM * K_PAD + OUT_DIM * OUT_DIM + 255) / 256, 256, 0, stream>>>(
        agg_W1, agg_W2, W1t, W2t);

    k_agg_gemm<<<B / 16, 256, 0, stream>>>(
        h_self, issuers, sectors, iss_sums, sec_sums,
        W1t, agg_b1, W2t, agg_b2, out);
}
